// GPT_14577119003069
// MI455X (gfx1250) — matrix operand order repaired and checked by reading, compiled, not run on hardware
//
#include <hip/hip_runtime.h>
#include <math.h>

// ---------------- problem constants (GPT-2 small, B=2, T=1024) ----------------
static constexpr int Bn   = 2;
static constexpr int Tn   = 1024;
static constexpr int Dn   = 768;
static constexpr int Hn   = 12;
static constexpr int HSn  = 64;
static constexpr int Ln   = 4;
static constexpr int Vn   = 50257;
static constexpr int DFFn = 3072;
static constexpr int Mn   = Bn * Tn; // 2048

// ---------------- vector types for WMMA ----------------
typedef __attribute__((ext_vector_type(16))) _Float16 v16h;
typedef __attribute__((ext_vector_type(8)))  _Float16 v8h;
typedef __attribute__((ext_vector_type(4)))  _Float16 v4h;
typedef __attribute__((ext_vector_type(8)))  float    v8f;

// ---------------- fragment loaders (wave32, 16x16x32 f16 WMMA) ----------------
// A (16x32, MxK): lanes 0-15 -> M=0..15, lanes 16-31 -> M=0..15 (other K half).
// half=lane>>4 selects K chunks {half*8..+7} (VGPR0-3) and {16+half*8..+7}
// (VGPR4-7): two contiguous 16B chunks per lane.
__device__ __forceinline__ v16h frag_a(const _Float16* rowp, int half) {
    union { v16h v; v8h h[2]; } u;
    u.h[0] = *(const v8h*)(rowp + (half << 3));
    u.h[1] = *(const v8h*)(rowp + 16 + (half << 3));
    return u.v;
}
// B (32x16, KxN) stored transposed in LDS as [n][k]: lane%16 -> N,
// half selects K chunks {half*8..+7} and {16+half*8..+7} (same as A).
__device__ __forceinline__ v16h frag_b(const _Float16* rowp, int half) {
    union { v16h v; v8h h[2]; } u;
    u.h[0] = *(const v8h*)(rowp + (half << 3));
    u.h[1] = *(const v8h*)(rowp + 16 + (half << 3));
    return u.v;
}

__device__ __forceinline__ v8f wmma16(v16h a, v16h b, v8f c) {
    return __builtin_amdgcn_wmma_f32_16x16x32_f16(false, a, false, b, (short)0, c, false, false);
}

__device__ __forceinline__ float gelu_tanh(float x) {
    return 0.5f * x * (1.0f + tanhf(0.7978845608028654f * (x + 0.044715f * x * x * x)));
}

// ---------------- WMMA GEMM: C[M,N] = act(A[M,K]@B[K,N] + bias) + resid -----
// 256 threads = 8 waves; block tile 128x128; wave tile 32x64 (2x4 WMMA frags).
// Staging software-pipelined through registers (next K-tile loads issued
// before the WMMA phase of the current tile).
// NEDGE=false: M,N exact multiples of the tile, fully vectorized, no guards.
// NEDGE=true : N ragged (LM head) -> clamped scalar B loads, guarded stores.
template <bool NEDGE>
__global__ __launch_bounds__(256) void gemm_f16_wmma(
    const float* __restrict__ A, const float* __restrict__ Bm,
    const float* __restrict__ bias, const float* __restrict__ resid,
    float* __restrict__ C, int M, int N, int K, int act)
{
    __shared__ __align__(32) _Float16 As[128][32];   // [m][k]  8 KB
    __shared__ __align__(32) _Float16 Bs[128][32];   // [n][k]  8 KB (transposed)

    const int tid  = threadIdx.x;
    const int lane = tid & 31;
    const int wave = tid >> 5;
    const int wm   = (wave >> 1) << 5;     // 0,32,64,96
    const int wn   = (wave &  1) << 6;     // 0 or 64
    const int m0   = blockIdx.y << 7;
    const int n0   = blockIdx.x << 7;
    const int lm   = lane & 15;
    const int half = lane >> 4;

    v8f acc[2][4];
    #pragma unroll
    for (int i = 0; i < 2; ++i)
        #pragma unroll
        for (int j = 0; j < 4; ++j) acc[i][j] = {};

    float4 ra[4];
    float4 rb4[4];
    float  rbs[16];

    auto load_a = [&](int kk) {
        #pragma unroll
        for (int p = 0; p < 4; ++p) {
            int idx4 = tid + (p << 8);           // 0..1023 float4s
            int r = idx4 >> 3;                   // row 0..127
            int c = (idx4 & 7) << 2;             // col 0,4,..,28
            ra[p] = *(const float4*)(A + (size_t)(m0 + r) * K + kk + c);
        }
    };
    auto load_b = [&](int kk) {
        if (!NEDGE) {
            #pragma unroll
            for (int p = 0; p < 4; ++p) {
                int idx4 = tid + (p << 8);       // 0..1023 float4s
                int r = idx4 >> 5;               // k 0..31
                int c = (idx4 & 31) << 2;        // n 0,4,..,124
                rb4[p] = *(const float4*)(Bm + (size_t)(kk + r) * N + n0 + c);
            }
        } else {
            #pragma unroll
            for (int p = 0; p < 16; ++p) {
                int idx = tid + (p << 8);        // 0..4095
                int r = idx >> 7;                // k 0..31
                int c = idx & 127;               // n 0..127
                int gn = n0 + c;
                gn = (gn < N) ? gn : (N - 1);    // branchless clamp
                rbs[p] = Bm[(size_t)(kk + r) * N + gn];
            }
        }
    };
    auto store_tiles = [&]() {
        #pragma unroll
        for (int p = 0; p < 4; ++p) {
            int idx4 = tid + (p << 8);
            int r = idx4 >> 3;
            int c = (idx4 & 7) << 2;
            v4h h = { (_Float16)ra[p].x, (_Float16)ra[p].y,
                      (_Float16)ra[p].z, (_Float16)ra[p].w };
            *(v4h*)(&As[r][c]) = h;
        }
        if (!NEDGE) {
            #pragma unroll
            for (int p = 0; p < 4; ++p) {
                int idx4 = tid + (p << 8);
                int r = idx4 >> 5;
                int c = (idx4 & 31) << 2;
                Bs[c + 0][r] = (_Float16)rb4[p].x;
                Bs[c + 1][r] = (_Float16)rb4[p].y;
                Bs[c + 2][r] = (_Float16)rb4[p].z;
                Bs[c + 3][r] = (_Float16)rb4[p].w;
            }
        } else {
            #pragma unroll
            for (int p = 0; p < 16; ++p) {
                int idx = tid + (p << 8);
                int r = idx >> 7;
                int c = idx & 127;
                Bs[c][r] = (_Float16)rbs[p];
            }
        }
    };

    load_a(0);
    load_b(0);

    for (int k0 = 0; k0 < K; k0 += 32) {
        store_tiles();                  // waits for in-flight loads, cvt -> LDS
        __syncthreads();
        if (k0 + 32 < K) {              // issue next tile's loads early
            load_a(k0 + 32);
            load_b(k0 + 32);
        }

        v16h a0 = frag_a(&As[wm + lm][0],      half);
        v16h a1 = frag_a(&As[wm + 16 + lm][0], half);
        v16h b0 = frag_b(&Bs[wn + lm][0],      half);
        v16h b1 = frag_b(&Bs[wn + 16 + lm][0], half);
        v16h b2 = frag_b(&Bs[wn + 32 + lm][0], half);
        v16h b3 = frag_b(&Bs[wn + 48 + lm][0], half);

        acc[0][0] = wmma16(a0, b0, acc[0][0]);
        acc[0][1] = wmma16(a0, b1, acc[0][1]);
        acc[0][2] = wmma16(a0, b2, acc[0][2]);
        acc[0][3] = wmma16(a0, b3, acc[0][3]);
        acc[1][0] = wmma16(a1, b0, acc[1][0]);
        acc[1][1] = wmma16(a1, b1, acc[1][1]);
        acc[1][2] = wmma16(a1, b2, acc[1][2]);
        acc[1][3] = wmma16(a1, b3, acc[1][3]);
        __syncthreads();
    }

    // C/D layout: lane%16 -> N, VGPR r -> M = (lane<16 ? r : 8+r)
    const int mo = half << 3;
    #pragma unroll
    for (int j = 0; j < 4; ++j) {
        const int gn = n0 + wn + (j << 4) + lm;
        if (NEDGE && gn >= N) continue;
        const float bv = bias ? bias[gn] : 0.0f;
        #pragma unroll
        for (int i = 0; i < 2; ++i) {
            const int mb = m0 + wm + (i << 4) + mo;
            #pragma unroll
            for (int r = 0; r < 8; ++r) {
                const int gm = mb + r;
                float v = acc[i][j][r] + bv;
                if (act) v = gelu_tanh(v);
                if (resid) v += resid[(size_t)gm * N + gn];
                C[(size_t)gm * N + gn] = v;
            }
        }
    }
}

// ---------------- WMMA flash attention ---------------------------------------
// Grid: (T/64, B*H); 128 threads = 4 waves; each wave owns a 16-query strip.
// Per 32-key tile: S = Q@K^T via 4 WMMAs, causal mask + online softmax with
// cross-lane (16-wide) shuffle reductions, P re-laid out C->A through a
// per-wave LDS tile, Y += P@V via 4 WMMAs.
__global__ __launch_bounds__(128) void attn_wmma_kernel(const float* __restrict__ qkv,
                                                        float* __restrict__ y)
{
    __shared__ __align__(32) _Float16 Ks[32][64];     // [key][d]   4 KB
    __shared__ __align__(32) _Float16 Vt[64][32];     // [d][key]   4 KB
    __shared__ __align__(32) _Float16 Pl[4][16][32];  // per-wave P 4 KB

    const int tid  = threadIdx.x;
    const int lane = tid & 31;
    const int wave = tid >> 5;
    const int lm   = lane & 15;
    const int half = lane >> 4;
    const int bh   = blockIdx.y;
    const int b    = bh / Hn;
    const int h    = bh % Hn;
    const int q0   = blockIdx.x << 6;          // workgroup query base
    const int qw   = q0 + (wave << 4);         // wave query base

    const size_t rs3 = (size_t)(3 * Dn);
    const float scale = 0.125f;                // 1/sqrt(64)

    // Q fragments (rows qw+lm), pre-scaled, A-frag element order
    v16h qf[2];
    {
        const float* qrow = qkv + (size_t)(b * Tn + qw + lm) * rs3 + h * HSn;
        #pragma unroll
        for (int c = 0; c < 2; ++c) {
            union { v16h v; _Float16 e[16]; } u;
            #pragma unroll
            for (int e = 0; e < 8; ++e)
                u.e[e] = (_Float16)(qrow[(c << 5) + (half << 3) + e] * scale);
            #pragma unroll
            for (int e = 0; e < 8; ++e)
                u.e[8 + e] = (_Float16)(qrow[(c << 5) + 16 + (half << 3) + e] * scale);
            qf[c] = u.v;
        }
    }

    v8f yacc[4];
    #pragma unroll
    for (int j = 0; j < 4; ++j) yacc[j] = {};
    float mrow[8], lrow[8];
    #pragma unroll
    for (int r = 0; r < 8; ++r) { mrow[r] = -3.0e38f; lrow[r] = 0.0f; }

    const int ntiles = (q0 >> 5) + 2;          // keys up to q0+63
    for (int kt = 0; kt < ntiles; ++kt) {
        const int kb = kt << 5;                // key tile base

        // stage K tile [32][64] row-major and V tile transposed [64][32]
        #pragma unroll
        for (int p = 0; p < 4; ++p) {
            int idx4 = tid + (p << 7);         // 0..511 float4s
            int r = idx4 >> 4;                 // key 0..31
            int c = (idx4 & 15) << 2;          // d 0,4,..,60
            const float* kr = qkv + (size_t)(b * Tn + kb + r) * rs3 + Dn + h * HSn;
            float4 v = *(const float4*)(kr + c);
            v4h hh = { (_Float16)v.x, (_Float16)v.y, (_Float16)v.z, (_Float16)v.w };
            *(v4h*)(&Ks[r][c]) = hh;
        }
        #pragma unroll
        for (int p = 0; p < 4; ++p) {
            int idx4 = tid + (p << 7);
            int r = idx4 >> 4;                 // key
            int c = (idx4 & 15) << 2;          // d
            const float* vr = qkv + (size_t)(b * Tn + kb + r) * rs3 + 2 * Dn + h * HSn;
            float4 v = *(const float4*)(vr + c);
            Vt[c + 0][r] = (_Float16)v.x;
            Vt[c + 1][r] = (_Float16)v.y;
            Vt[c + 2][r] = (_Float16)v.z;
            Vt[c + 3][r] = (_Float16)v.w;
        }
        __syncthreads();

        if (kb <= qw + 15) {                   // wave has keys in this tile
            // scores: two 16x16 tiles (keys kb..+15, kb+16..+31)
            v8f s0 = {}, s1 = {};
            #pragma unroll
            for (int c = 0; c < 2; ++c) {
                v16h k0 = frag_b(&Ks[lm][c << 5],      half);
                v16h k1 = frag_b(&Ks[16 + lm][c << 5], half);
                s0 = wmma16(qf[c], k0, s0);
                s1 = wmma16(qf[c], k1, s1);
            }
            // causal mask + per-row stats (C layout: lane=col, r -> row mo+r)
            const int kc0 = kb + lm, kc1 = kb + 16 + lm;
            float p0[8], p1[8], pm[8];
            #pragma unroll
            for (int r = 0; r < 8; ++r) {
                int qg = qw + (half << 3) + r;
                p0[r] = (kc0 <= qg) ? s0[r] : -3.0e38f;
                p1[r] = (kc1 <= qg) ? s1[r] : -3.0e38f;
                pm[r] = fmaxf(p0[r], p1[r]);
            }
            #pragma unroll
            for (int r = 0; r < 8; ++r) {      // row max over 16 lanes
                float v = pm[r];
                v = fmaxf(v, __shfl_xor(v, 1, 16));
                v = fmaxf(v, __shfl_xor(v, 2, 16));
                v = fmaxf(v, __shfl_xor(v, 4, 16));
                v = fmaxf(v, __shfl_xor(v, 8, 16));
                pm[r] = v;
            }
            float corr[8];
            #pragma unroll
            for (int r = 0; r < 8; ++r) {
                float mnew = fmaxf(mrow[r], pm[r]);
                corr[r] = __expf(mrow[r] - mnew);
                p0[r] = __expf(p0[r] - mnew);
                p1[r] = __expf(p1[r] - mnew);
                mrow[r] = mnew;
            }
            #pragma unroll
            for (int r = 0; r < 8; ++r) {      // row sum over 16 lanes
                float v = p0[r] + p1[r];
                v += __shfl_xor(v, 1, 16);
                v += __shfl_xor(v, 2, 16);
                v += __shfl_xor(v, 4, 16);
                v += __shfl_xor(v, 8, 16);
                lrow[r] = lrow[r] * corr[r] + v;
            }
            #pragma unroll
            for (int j = 0; j < 4; ++j)        // rescale running output
                #pragma unroll
                for (int r = 0; r < 8; ++r) yacc[j][r] *= corr[r];
            // P: C layout -> A fragment via per-wave LDS tile
            #pragma unroll
            for (int r = 0; r < 8; ++r) {
                Pl[wave][(half << 3) + r][lm]      = (_Float16)p0[r];
                Pl[wave][(half << 3) + r][16 + lm] = (_Float16)p1[r];
            }
            v16h pA = frag_a(&Pl[wave][lm][0], half);
            #pragma unroll
            for (int j = 0; j < 4; ++j) {
                v16h vf = frag_b(&Vt[(j << 4) + lm][0], half);
                yacc[j] = wmma16(pA, vf, yacc[j]);
            }
        }
        __syncthreads();
    }

    // normalize and store y[B*T, D] head-major
    #pragma unroll
    for (int j = 0; j < 4; ++j) {
        #pragma unroll
        for (int r = 0; r < 8; ++r) {
            int q = qw + (half << 3) + r;
            y[(size_t)(b * Tn + q) * Dn + h * HSn + (j << 4) + lm] = yacc[j][r] / lrow[r];
        }
    }
}

// ---------------- embedding: x = wte[tok] + wpe[t] ----------------
__global__ void embed_kernel(const int* __restrict__ tokens, const float* __restrict__ wte,
                             const float* __restrict__ wpe, float* __restrict__ x)
{
    int idx = blockIdx.x * blockDim.x + threadIdx.x;
    if (idx >= Mn * Dn) return;
    int d    = idx % Dn;
    int bt   = idx / Dn;        // 0 .. Mn-1
    int tpos = bt % Tn;
    int tok  = tokens[bt];
    x[idx] = wte[(size_t)tok * Dn + d] + wpe[(size_t)tpos * Dn + d];
}

// ---------------- LayerNorm: one 256-thread block per row ----------------
__global__ __launch_bounds__(256) void layernorm_kernel(
    const float* __restrict__ x, const float* __restrict__ w,
    const float* __restrict__ b, float* __restrict__ out, int D)
{
    __shared__ float red[256];
    const int row = blockIdx.x;
    const int tid = threadIdx.x;
    const float* xr = x + (size_t)row * D;

    float s = 0.0f;
    for (int i = tid; i < D; i += 256) s += xr[i];
    red[tid] = s; __syncthreads();
    for (int off = 128; off > 0; off >>= 1) {
        if (tid < off) red[tid] += red[tid + off];
        __syncthreads();
    }
    float mean = red[0] / (float)D;
    __syncthreads();

    float vs = 0.0f;
    for (int i = tid; i < D; i += 256) { float dv = xr[i] - mean; vs += dv * dv; }
    red[tid] = vs; __syncthreads();
    for (int off = 128; off > 0; off >>= 1) {
        if (tid < off) red[tid] += red[tid + off];
        __syncthreads();
    }
    float inv = rsqrtf(red[0] / (float)D + 1e-5f);

    for (int i = tid; i < D; i += 256)
        out[(size_t)row * D + i] = (xr[i] - mean) * inv * w[i] + b[i];
}

// ---------------- driver ----------------
extern "C" void kernel_launch(void* const* d_in, const int* in_sizes, int n_in,
                              void* d_out, int out_size, void* d_ws, size_t ws_size,
                              hipStream_t stream)
{
    (void)in_sizes; (void)n_in; (void)out_size; (void)ws_size;

    const int*   tokens = (const int*)  d_in[0];
    const float* wte  = (const float*)d_in[1];
    const float* wpe  = (const float*)d_in[2];
    const float* ln1w = (const float*)d_in[3];
    const float* ln1b = (const float*)d_in[4];
    const float* wqkv = (const float*)d_in[5];
    const float* bqkv = (const float*)d_in[6];
    const float* wo   = (const float*)d_in[7];
    const float* bo   = (const float*)d_in[8];
    const float* ln2w = (const float*)d_in[9];
    const float* ln2b = (const float*)d_in[10];
    const float* wfc  = (const float*)d_in[11];
    const float* bfc  = (const float*)d_in[12];
    const float* wpr  = (const float*)d_in[13];
    const float* bpr  = (const float*)d_in[14];
    const float* lnfw = (const float*)d_in[15];
    const float* lnfb = (const float*)d_in[16];
    const float* lmw  = (const float*)d_in[17];

    // workspace layout (floats): x | h | qkv | y | mlp
    float* x    = (float*)d_ws;
    float* hbuf = x    + (size_t)Mn * Dn;
    float* qkv  = hbuf + (size_t)Mn * Dn;
    float* ybuf = qkv  + (size_t)Mn * 3 * Dn;
    float* mbuf = ybuf + (size_t)Mn * Dn;

    embed_kernel<<<(Mn * Dn + 255) / 256, 256, 0, stream>>>(tokens, wte, wpe, x);

    for (int l = 0; l < Ln; ++l) {
        layernorm_kernel<<<Mn, 256, 0, stream>>>(x, ln1w + l * Dn, ln1b + l * Dn, hbuf, Dn);
        // qkv = h @ wqkv + bqkv          [2048,768]x[768,2304]
        gemm_f16_wmma<false><<<dim3(3 * Dn / 128, Mn / 128), 256, 0, stream>>>(
            hbuf, wqkv + (size_t)l * Dn * 3 * Dn, bqkv + (size_t)l * 3 * Dn,
            nullptr, qkv, Mn, 3 * Dn, Dn, 0);
        attn_wmma_kernel<<<dim3(Tn / 64, Bn * Hn), 128, 0, stream>>>(qkv, ybuf);
        // x = x + y @ wo + bo            [2048,768]x[768,768]
        gemm_f16_wmma<false><<<dim3(Dn / 128, Mn / 128), 256, 0, stream>>>(
            ybuf, wo + (size_t)l * Dn * Dn, bo + (size_t)l * Dn,
            x, x, Mn, Dn, Dn, 0);
        layernorm_kernel<<<Mn, 256, 0, stream>>>(x, ln2w + l * Dn, ln2b + l * Dn, hbuf, Dn);
        // m = gelu(h @ wfc + bfc)        [2048,768]x[768,3072]
        gemm_f16_wmma<false><<<dim3(DFFn / 128, Mn / 128), 256, 0, stream>>>(
            hbuf, wfc + (size_t)l * Dn * DFFn, bfc + (size_t)l * DFFn,
            nullptr, mbuf, Mn, DFFn, Dn, 1);
        // x = x + m @ wpr + bpr          [2048,3072]x[3072,768]
        gemm_f16_wmma<false><<<dim3(Dn / 128, Mn / 128), 256, 0, stream>>>(
            mbuf, wpr + (size_t)l * DFFn * Dn, bpr + (size_t)l * Dn,
            x, x, Mn, Dn, DFFn, 0);
    }

    layernorm_kernel<<<Mn, 256, 0, stream>>>(x, lnfw, lnfb, hbuf, Dn);
    // logits = h @ lm_w                  [2048,768]x[768,50257] (ragged N)
    gemm_f16_wmma<true><<<dim3((Vn + 127) / 128, Mn / 128), 256, 0, stream>>>(
        hbuf, lmw, nullptr, nullptr, (float*)d_out, Mn, Vn, Dn, 0);
}
